// RPEMultiHeadAttention_55018531062129
// MI455X (gfx1250) — matrix operand order repaired and checked by reading, not compiled, not run on hardware
//
#include <hip/hip_runtime.h>
#include <hip/hip_bf16.h>

// ---- types -----------------------------------------------------------------
typedef __attribute__((ext_vector_type(16))) _Float16 v16h;
typedef __attribute__((ext_vector_type(8)))  _Float16 v8h;
typedef __attribute__((ext_vector_type(2)))  __fp16   c2h;   // cvt_pkrtz result type
typedef __attribute__((ext_vector_type(8)))  float    v8f;
typedef __attribute__((ext_vector_type(4)))  float    f32x4;

#define DM   256   // D_MODEL
#define NN   768   // N == M
#define NH   4     // heads
#define HD   64    // head dim
#define NM   589824   // 768*768

union V16H { v16h v; v8h h8[2]; c2h p2[8]; };

__device__ __forceinline__ v8f wmma16(v16h a, v16h b, v8f c) {
  // D = A(16x32 f16) * B(32x16 f16) + C(16x16 f32)
  return __builtin_amdgcn_wmma_f32_16x16x32_f16(false, a, false, b, (short)0, c,
                                                false, false);
}

// A operand, f16 row-major [row][k]; lane r=L&15 holds row m0+r,
// halves 0..15 = K[hi*16 .. +16) (contiguous)
__device__ __forceinline__ v16h ldA_h(const _Float16* __restrict__ A, int lda,
                                      int m0, int k0, int lane) {
  int r = lane & 15, hi = lane >> 4;
  const _Float16* p = A + (size_t)(m0 + r) * lda + k0;
  V16H u;
  u.h8[0] = *(const v8h*)(p + hi * 16);
  u.h8[1] = *(const v8h*)(p + 8 + hi * 16);
  return u.v;
}

// B^T operand, f16 row-major [j][k]; lane c=L&15 holds column j0+c,
// halves 0..15 = K[hi*16 .. +16) (contiguous)
__device__ __forceinline__ v16h ldB_h(const _Float16* __restrict__ B, int ldb,
                                      int j0, int k0, int lane) {
  int c = lane & 15, hi = lane >> 4;
  const _Float16* p = B + (size_t)(j0 + c) * ldb + k0 + hi * 16;
  V16H u;
  u.h8[0] = *(const v8h*)p;
  u.h8[1] = *(const v8h*)(p + 8);
  return u.v;
}

// A operand loaded from f32 memory, converted inline with v_cvt_pk_rtz_f16_f32
__device__ __forceinline__ v16h ldA_f32(const float* __restrict__ A, int lda,
                                        int m0, int k0, int lane) {
  int r = lane & 15, hi = lane >> 4;
  const float* p = A + (size_t)(m0 + r) * lda + k0 + hi * 16;
  f32x4 a0 = *(const f32x4*)(p);
  f32x4 a1 = *(const f32x4*)(p + 4);
  f32x4 a2 = *(const f32x4*)(p + 8);
  f32x4 a3 = *(const f32x4*)(p + 12);
  V16H u;
  u.p2[0] = __builtin_amdgcn_cvt_pkrtz(a0.x, a0.y);
  u.p2[1] = __builtin_amdgcn_cvt_pkrtz(a0.z, a0.w);
  u.p2[2] = __builtin_amdgcn_cvt_pkrtz(a1.x, a1.y);
  u.p2[3] = __builtin_amdgcn_cvt_pkrtz(a1.z, a1.w);
  u.p2[4] = __builtin_amdgcn_cvt_pkrtz(a2.x, a2.y);
  u.p2[5] = __builtin_amdgcn_cvt_pkrtz(a2.z, a2.w);
  u.p2[6] = __builtin_amdgcn_cvt_pkrtz(a3.x, a3.y);
  u.p2[7] = __builtin_amdgcn_cvt_pkrtz(a3.z, a3.w);
  return u.v;
}

__device__ __forceinline__ v16h ldB_f32(const float* __restrict__ B, int ldb,
                                        int j0, int k0, int lane) {
  int c = lane & 15, hi = lane >> 4;
  const float* p = B + (size_t)(j0 + c) * ldb + k0 + hi * 16;
  f32x4 a0 = *(const f32x4*)(p);
  f32x4 a1 = *(const f32x4*)(p + 4);
  f32x4 a2 = *(const f32x4*)(p + 8);
  f32x4 a3 = *(const f32x4*)(p + 12);
  V16H u;
  u.p2[0] = __builtin_amdgcn_cvt_pkrtz(a0.x, a0.y);
  u.p2[1] = __builtin_amdgcn_cvt_pkrtz(a0.z, a0.w);
  u.p2[2] = __builtin_amdgcn_cvt_pkrtz(a1.x, a1.y);
  u.p2[3] = __builtin_amdgcn_cvt_pkrtz(a1.z, a1.w);
  u.p2[4] = __builtin_amdgcn_cvt_pkrtz(a2.x, a2.y);
  u.p2[5] = __builtin_amdgcn_cvt_pkrtz(a2.z, a2.w);
  u.p2[6] = __builtin_amdgcn_cvt_pkrtz(a3.x, a3.y);
  u.p2[7] = __builtin_amdgcn_cvt_pkrtz(a3.z, a3.w);
  return u.v;
}

// ---- prep: WpT (f16, [e][d] = Wp[d][e]) ------------------------------------
__global__ void k_wpT(const float* __restrict__ Wp, _Float16* __restrict__ WpT) {
  int t = blockIdx.x * 256 + threadIdx.x;   // 65536
  int e = t >> 8, d = t & 255;
  WpT[e * DM + d] = (_Float16)Wp[d * DM + e];
}

__global__ void k_zero_h(_Float16* __restrict__ p, int n) {
  int t = blockIdx.x * 256 + threadIdx.x;
  if (t < n) p[t] = (_Float16)0.0f;
}

// ---- QKV projection: Out = X @ W^T + b  (768x256) --------------------------
// isV==0: store f16 row-major to OutH;  isV==1: store f32 to Vf
__global__ __launch_bounds__(32) void k_proj(const float* __restrict__ X,
                                             const float* __restrict__ W,
                                             const float* __restrict__ bias,
                                             _Float16* __restrict__ OutH,
                                             float* __restrict__ Vf, int isV) {
  int lane = threadIdx.x;
  int m0 = blockIdx.x * 16, j0 = blockIdx.y * 16;
  v8f acc0 = {0, 0, 0, 0, 0, 0, 0, 0}, acc1 = {0, 0, 0, 0, 0, 0, 0, 0};
#pragma unroll
  for (int k0 = 0; k0 < DM; k0 += 64) {
    acc0 = wmma16(ldA_f32(X, DM, m0, k0, lane),      ldB_f32(W, DM, j0, k0, lane),      acc0);
    acc1 = wmma16(ldA_f32(X, DM, m0, k0 + 32, lane), ldB_f32(W, DM, j0, k0 + 32, lane), acc1);
  }
  int col = j0 + (lane & 15), hi = lane >> 4;
  float bc = bias[col];
#pragma unroll
  for (int i = 0; i < 8; ++i) {
    float v = acc0[i] + acc1[i] + bc;
    int row = m0 + hi * 8 + i;
    if (isV) Vf[(size_t)row * DM + col] = v;
    else     OutH[(size_t)row * DM + col] = (_Float16)v;
  }
}

// ---- qb[h][n] = q_h[n,:] . bp_h --------------------------------------------
__global__ void k_qb(const _Float16* __restrict__ Qh, const float* __restrict__ bp,
                     float* __restrict__ qb) {
  int t = blockIdx.x * 256 + threadIdx.x;  // 3072
  if (t >= NH * NN) return;
  int h = t / NN, n = t % NN;
  float s = 0.f;
#pragma unroll 8
  for (int c = 0; c < HD; ++c) s += (float)Qh[n * DM + h * HD + c] * bp[h * HD + c];
  qb[h * NN + n] = s;
}

// ---- wqp: per head, Q_h(768x64) @ Wp_h(64x256) -> wqpB[n][h(pad16)][e] f16 --
__global__ __launch_bounds__(32) void k_wqp(const _Float16* __restrict__ Qh,
                                            const _Float16* __restrict__ WpT,
                                            _Float16* __restrict__ wqpB) {
  int lane = threadIdx.x;
  int n0 = blockIdx.x * 16, e0 = blockIdx.y * 16, h = blockIdx.z;
  v8f acc = {0, 0, 0, 0, 0, 0, 0, 0};
  acc = wmma16(ldA_h(Qh + h * HD, DM, n0, 0, lane),  ldB_h(WpT + h * HD, DM, e0, 0, lane),  acc);
  acc = wmma16(ldA_h(Qh + h * HD, DM, n0, 32, lane), ldB_h(WpT + h * HD, DM, e0, 32, lane), acc);
  int e = e0 + (lane & 15), hi = lane >> 4;
#pragma unroll
  for (int i = 0; i < 8; ++i) {
    int n = n0 + hi * 8 + i;
    wqpB[(size_t)n * (16 * DM) + h * DM + e] = (_Float16)acc[i];
  }
}

// ---- scores_e[h][n][m] = q_h[n] . k_h[m] -----------------------------------
__global__ __launch_bounds__(32) void k_se(const _Float16* __restrict__ Qh,
                                           const _Float16* __restrict__ Kh,
                                           float* __restrict__ Se) {
  int lane = threadIdx.x;
  int n0 = blockIdx.x * 16, m0 = blockIdx.y * 16, h = blockIdx.z;
  v8f acc = {0, 0, 0, 0, 0, 0, 0, 0};
  acc = wmma16(ldA_h(Qh + h * HD, DM, n0, 0, lane),  ldB_h(Kh + h * HD, DM, m0, 0, lane),  acc);
  acc = wmma16(ldA_h(Qh + h * HD, DM, n0, 32, lane), ldB_h(Kh + h * HD, DM, m0, 32, lane), acc);
  int m = m0 + (lane & 15), hi = lane >> 4;
#pragma unroll
  for (int i = 0; i < 8; ++i) {
    int n = n0 + hi * 8 + i;
    Se[(size_t)h * NM + (size_t)n * NN + m] = acc[i];
  }
}

// ---- main stream: scores_p via WMMA + fused softmax -> attn ----------------
// one block per n; 8 waves; streams E[n] (768x256 f32) exactly once
__global__ __launch_bounds__(256) void k_scores(const float* __restrict__ E,
                                                const _Float16* __restrict__ wqpB,
                                                const float* __restrict__ Se,
                                                const float* __restrict__ qb,
                                                float* __restrict__ attn) {
  __shared__ float ss[NN * 4];
  int n = blockIdx.x;
  int tid = threadIdx.x, lane = tid & 31, w = tid >> 5;
  int r = lane & 15, hi = lane >> 4;
  const float* En = E + (size_t)n * NN * DM;

  // preload all 8 B operands (j = head row, k = e): 64 VGPRs, reused by 6 tiles
  const _Float16* Bn = wqpB + (size_t)n * (16 * DM);
  v16h bmat[8];
#pragma unroll
  for (int i = 0; i < 8; ++i) {
    const _Float16* p = Bn + r * DM + i * 32 + hi * 16;
    V16H u; u.h8[0] = *(const v8h*)p; u.h8[1] = *(const v8h*)(p + 8);
    bmat[i] = u.v;
  }
  float qbl = (r < 4) ? qb[r * NN + n] : 0.f;

  for (int t = w; t < 48; t += 8) {
    int m0 = t * 16;
    v8f a0 = {0, 0, 0, 0, 0, 0, 0, 0}, a1 = {0, 0, 0, 0, 0, 0, 0, 0};
#pragma unroll
    for (int i = 0; i < 8; i += 2) {
      a0 = wmma16(ldA_f32(En, DM, m0, i * 32, lane),       bmat[i],     a0);
      a1 = wmma16(ldA_f32(En, DM, m0, (i + 1) * 32, lane), bmat[i + 1], a1);
    }
    if (r < 4) {
#pragma unroll
      for (int i = 0; i < 8; ++i) {
        int m = m0 + hi * 8 + i;
        float v = (a0[i] + a1[i] + Se[(size_t)r * NM + (size_t)n * NN + m] + qbl) * 0.125f;
        ss[m * 4 + r] = v;
      }
    }
  }
  __syncthreads();

  if (w < 4) {  // one wave per head: softmax over m
    int h = w;
    float mx = -3.0e38f;
    for (int m = lane; m < NN; m += 32) mx = fmaxf(mx, ss[m * 4 + h]);
#pragma unroll
    for (int o = 16; o; o >>= 1) mx = fmaxf(mx, __shfl_xor(mx, o, 32));
    float sum = 0.f;
    for (int m = lane; m < NN; m += 32) {
      float e = __expf(ss[m * 4 + h] - mx);
      ss[m * 4 + h] = e;
      sum += e;
    }
#pragma unroll
    for (int o = 16; o; o >>= 1) sum += __shfl_xor(sum, o, 32);
    float inv = 1.f / sum;
    float* out = attn + (size_t)h * NM + (size_t)n * NN;
    for (int m = lane; m < NN; m += 32) out[m] = ss[m * 4 + h] * inv;
  }
}

// ---- hidden = attn @ V  (exact fp32; V resident in L2) ---------------------
__global__ __launch_bounds__(256) void k_av(const float* __restrict__ attn,
                                            const float* __restrict__ Vf,
                                            float* __restrict__ hidden) {
  int n = blockIdx.x, d = threadIdx.x, h = d >> 6;
  const float* ar = attn + (size_t)h * NM + (size_t)n * NN;
  float s = 0.f;
#pragma unroll 4
  for (int m = 0; m < NN; m += 4) {
    s += ar[m]     * Vf[(size_t)(m)     * DM + d];
    s += ar[m + 1] * Vf[(size_t)(m + 1) * DM + d];
    s += ar[m + 2] * Vf[(size_t)(m + 2) * DM + d];
    s += ar[m + 3] * Vf[(size_t)(m + 3) * DM + d];
  }
  hidden[(size_t)n * DM + d] = s;
}

// ---- launcher --------------------------------------------------------------
extern "C" void kernel_launch(void* const* d_in, const int* in_sizes, int n_in,
                              void* d_out, int out_size, void* d_ws, size_t ws_size,
                              hipStream_t stream) {
  const float* input_q  = (const float*)d_in[0];
  const float* input_k  = (const float*)d_in[1];
  const float* input_v  = (const float*)d_in[2];
  const float* embed_qk = (const float*)d_in[3];
  const float* Wq = (const float*)d_in[4];
  const float* bq = (const float*)d_in[5];
  const float* Wk = (const float*)d_in[6];
  const float* bk = (const float*)d_in[7];
  const float* Wv = (const float*)d_in[8];
  const float* bv = (const float*)d_in[9];
  const float* Wp = (const float*)d_in[10];
  const float* bp = (const float*)d_in[11];

  char* ws = (char*)d_ws;
  _Float16* Qh   = (_Float16*)(ws + 0);         // 768*256 f16
  _Float16* Kh   = (_Float16*)(ws + 393216);    // 768*256 f16
  float*    Vf   = (float*)   (ws + 786432);    // 768*256 f32
  _Float16* WpT  = (_Float16*)(ws + 1572864);   // 256*256 f16
  _Float16* wqpB = (_Float16*)(ws + 1703936);   // 768*16*256 f16
  float*    qb   = (float*)   (ws + 7995392);   // 4*768 f32
  float*    Se   = (float*)   (ws + 8007680);   // 4*768*768 f32

  float* hidden = (float*)d_out;
  float* attn   = (float*)d_out + NN * DM;      // hidden first, then attn

  k_wpT   <<<256, 256, 0, stream>>>(Wp, WpT);
  k_zero_h<<<12288, 256, 0, stream>>>(wqpB, NN * 16 * DM);

  k_proj<<<dim3(48, 16), 32, 0, stream>>>(input_q, Wq, bq, Qh, Vf, 0);
  k_proj<<<dim3(48, 16), 32, 0, stream>>>(input_k, Wk, bk, Kh, Vf, 0);
  k_proj<<<dim3(48, 16), 32, 0, stream>>>(input_v, Wv, bv, Qh, Vf, 1);

  k_qb <<<12, 256, 0, stream>>>(Qh, bp, qb);
  k_wqp<<<dim3(48, 16, 4), 32, 0, stream>>>(Qh, WpT, wqpB);
  k_se <<<dim3(48, 48, 4), 32, 0, stream>>>(Qh, Kh, Se);

  k_scores<<<NN, 256, 0, stream>>>(embed_qk, wqpB, Se, qb, attn);
  k_av    <<<NN, 256, 0, stream>>>(attn, Vf, hidden);
}
